// JointGeometryBlock_9964324127133
// MI455X (gfx1250) — matrix operand order repaired and checked by reading, not compiled, not run on hardware
//
#include <hip/hip_runtime.h>
#include <hip/hip_bf16.h>
#include <math.h>

// ---------------- fixed problem shape ----------------
static constexpr int Bv  = 2;
static constexpr int Tv  = 2048;
static constexpr int Cv  = 1024;
static constexpr int Hv  = 16;
static constexpr int HDv = 64;
static constexpr int NT  = Bv * Tv;          // 4096 tokens

typedef __attribute__((ext_vector_type(16))) _Float16 v16h;
typedef __attribute__((ext_vector_type(8)))  _Float16 v8h;
typedef __attribute__((ext_vector_type(8)))  float    v8f;

// ---------------- CDNA5 async copy: global -> LDS (ASYNCcnt tracked) -------
// builtin signature (from probe): (int4 AS1*, int4 AS3*, imm offset, imm cpol)
typedef int v4i_vs __attribute__((vector_size(16)));
typedef __attribute__((address_space(1))) v4i_vs* gv4i_p;
typedef __attribute__((address_space(3))) v4i_vs* lv4i_p;

__device__ __forceinline__ void async_copy16(const _Float16* g, _Float16* l) {
#if __has_builtin(__builtin_amdgcn_global_load_async_to_lds_b128)
  __builtin_amdgcn_global_load_async_to_lds_b128((gv4i_p)g, (lv4i_p)l, 0, 0);
#else
  *(v8h*)l = *(const v8h*)g;   // synchronous fallback
#endif
}

template <int N>
__device__ __forceinline__ void wait_async() {
#if __has_builtin(__builtin_amdgcn_s_wait_asynccnt)
  __builtin_amdgcn_s_wait_asynccnt(N);
#else
  asm volatile("s_wait_asynccnt %0" ::"i"(N) : "memory");
#endif
}

// ---------------- WMMA fragment helpers (CDNA5 16x16x32 f16 layouts) -------
// A (16x32, MxK): lane m=lane&15, sel=lane>>4.
//   halves[0..7]  = A[m][8*sel .. +7], halves[8..15] = A[m][16 + 8*sel .. +7]
__device__ __forceinline__ v16h load_a_frag(const _Float16* __restrict__ A, int lda) {
  int lane = threadIdx.x & 31;
  int m = lane & 15, s = lane >> 4;
  const _Float16* p = A + (size_t)m * lda + 8 * s;
  v8h lo = *(const v8h*)p;
  v8h hi = *(const v8h*)(p + 16);
  v16h r;
#pragma unroll
  for (int i = 0; i < 8; ++i) { r[i] = lo[i]; r[i + 8] = hi[i]; }
  return r;
}

// B (32x16, KxN) sourced from row-major W[N][K]: lane n=lane&15, sel=lane>>4.
//   halves[0..7]  = W[n][8*sel .. +7], halves[8..15] = W[n][16 + 8*sel .. +7]
__device__ __forceinline__ v16h load_b_frag(const _Float16* __restrict__ W, int ldw) {
  int lane = threadIdx.x & 31;
  int n = lane & 15, s = lane >> 4;
  const _Float16* p = W + (size_t)n * ldw + 8 * s;
  v8h lo = *(const v8h*)p;
  v8h hi = *(const v8h*)(p + 16);
  v16h r;
#pragma unroll
  for (int i = 0; i < 8; ++i) { r[i] = lo[i]; r[i + 8] = hi[i]; }
  return r;
}

// ---------------- rmsnorm (f32 in -> f16 out) ----------------
__global__ __launch_bounds__(256) void rmsnorm_f16_kernel(
    const float* __restrict__ x, _Float16* __restrict__ out) {
  int row = blockIdx.x;
  const float* xr = x + (size_t)row * Cv;
  float ss = 0.f;
#pragma unroll
  for (int i = threadIdx.x; i < Cv; i += 256) { float v = xr[i]; ss += v * v; }
#pragma unroll
  for (int o = 16; o > 0; o >>= 1) ss += __shfl_xor(ss, o, 32);
  __shared__ float red[8];
  if ((threadIdx.x & 31) == 0) red[threadIdx.x >> 5] = ss;
  __syncthreads();
  float tot = 0.f;
#pragma unroll
  for (int i = 0; i < 8; ++i) tot += red[i];
  float scale = rsqrtf(tot * (1.0f / Cv) + 1e-6f);
  _Float16* orow = out + (size_t)row * Cv;
  for (int i = threadIdx.x; i < Cv; i += 256) orow[i] = (_Float16)(xr[i] * scale);
}

// ---------------- f32 -> f16 convert ----------------
__global__ __launch_bounds__(256) void f32_to_f16_kernel(
    const float* __restrict__ in, _Float16* __restrict__ out, int n) {
  int i = blockIdx.x * 256 + threadIdx.x;
  if (i < n) out[i] = (_Float16)in[i];
}

// ---------------- WMMA GEMM with async double-buffered LDS staging --------
// out = A[M,K] * W[N,K]^T (+bias)(+resid); block 256 thr (8 waves),
// block tile 128x128, wave tile 64x32, K step 32.
static constexpr int LP = 40;   // LDS row pitch in halves (80B: bank-conflict-free)

__global__ __launch_bounds__(256) void gemm_wmma_kernel(
    const _Float16* __restrict__ A, const _Float16* __restrict__ W,
    int M, int N, int K,
    const float* __restrict__ bias, const float* __restrict__ resid,
    float* __restrict__ outF, _Float16* __restrict__ outH) {
  __shared__ __align__(16) _Float16 shA[2][128][LP];
  __shared__ __align__(16) _Float16 shB[2][128][LP];

  int tiles_n = N >> 7;
  int bm = (blockIdx.x / tiles_n) << 7;
  int bn = (blockIdx.x % tiles_n) << 7;
  int wave = threadIdx.x >> 5;
  int lane = threadIdx.x & 31;
  int waveMl = (wave >> 2) << 6;   // 0 / 64
  int waveNl = (wave & 3) << 5;    // 0..96

  // stage one 128x32 A tile + 128x32 B tile (4 async b128 per thread)
  auto stage = [&](int buf, int k0) {
    int t = threadIdx.x;
#pragma unroll
    for (int i = 0; i < 2; ++i) {
      int idx = t + i * 256;          // 512 16B-transfers per tile pair
      int row = idx >> 2, seg = idx & 3;
      async_copy16(A + (size_t)(bm + row) * K + k0 + seg * 8, &shA[buf][row][seg * 8]);
      async_copy16(W + (size_t)(bn + row) * K + k0 + seg * 8, &shB[buf][row][seg * 8]);
    }
  };

  v8f acc[4][2] = {};
  int nk = K >> 5;
  stage(0, 0);
  for (int kt = 0; kt < nk; ++kt) {
    int cur = kt & 1;
    bool more = (kt + 1 < nk);
    if (more) stage(cur ^ 1, (kt + 1) << 5);   // overlap next tile with compute
    if (more) wait_async<4>(); else wait_async<0>();
    __syncthreads();                            // all waves' cur-tiles in LDS
    v16h bfr0 = load_b_frag(&shB[cur][waveNl][0], LP);
    v16h bfr1 = load_b_frag(&shB[cur][waveNl + 16][0], LP);
#pragma unroll
    for (int i = 0; i < 4; ++i) {
      v16h afr = load_a_frag(&shA[cur][waveMl + i * 16][0], LP);
      acc[i][0] = __builtin_amdgcn_wmma_f32_16x16x32_f16(
          false, afr, false, bfr0, (short)0, acc[i][0], false, false);
      acc[i][1] = __builtin_amdgcn_wmma_f32_16x16x32_f16(
          false, afr, false, bfr1, (short)0, acc[i][1], false, false);
    }
    __syncthreads();                            // done with cur before re-stage
  }

  int s = lane >> 4, n = lane & 15;
#pragma unroll
  for (int i = 0; i < 4; ++i)
#pragma unroll
    for (int j2 = 0; j2 < 2; ++j2) {
      int col = bn + waveNl + j2 * 16 + n;
      float bv = bias ? bias[col] : 0.f;
#pragma unroll
      for (int j = 0; j < 8; ++j) {
        int row = bm + waveMl + i * 16 + j + 8 * s;
        float v = acc[i][j2][j] + bv;
        size_t idx = (size_t)row * N + col;
        if (resid) v += resid[idx];
        if (outF) outF[idx] = v;
        if (outH) outH[idx] = (_Float16)v;
      }
    }
}

// ---------------- rotary + lorentz prefactor + V transpose ----------------
__global__ __launch_bounds__(256) void rotary_pack_kernel(
    const _Float16* __restrict__ qkv, const float* __restrict__ hyp,
    _Float16* __restrict__ Q, _Float16* __restrict__ Kd, _Float16* __restrict__ VT,
    float* __restrict__ q0, float* __restrict__ k0) {
  int idx = blockIdx.x * 256 + threadIdx.x;     // (b*T+t)*H + h
  int h = idx & (Hv - 1);
  int bt = idx >> 4;
  int t = bt & (Tv - 1);
  int b = bt >> 11;
  const _Float16* row = qkv + (size_t)bt * (3 * Cv);
  int bh = b * Hv + h;
  _Float16* Qrow = Q + ((size_t)bh * Tv + t) * HDv;
  _Float16* Krow = Kd + ((size_t)bh * Tv + t) * HDv;
  float kc = hyp[h];
  float q0s = kc, k0s = kc;
#pragma unroll 4
  for (int d = 0; d < HDv / 2; ++d) {
    float inv = powf(10000.0f, -(float)d * (2.0f / HDv));
    float fr = (float)t * inv;
    float cs = cosf(fr), sn = sinf(fr);
    float qa = (float)row[h * HDv + d], qb = (float)row[h * HDv + d + 32];
    float ka = (float)row[Cv + h * HDv + d], kb = (float)row[Cv + h * HDv + d + 32];
    float q1 = qa * cs + qb * sn, q2 = qb * cs - qa * sn;
    float k1 = ka * cs + kb * sn, k2 = kb * cs - ka * sn;
    q0s += q1 * q1 + q2 * q2;
    k0s += k1 * k1 + k2 * k2;
    Qrow[d] = (_Float16)q1; Qrow[d + 32] = (_Float16)q2;
    Krow[d] = (_Float16)k1; Krow[d + 32] = (_Float16)k2;
  }
#pragma unroll 4
  for (int d = 0; d < HDv; ++d)
    VT[((size_t)bh * HDv + d) * Tv + t] = row[2 * Cv + h * HDv + d];
  q0[(size_t)bh * Tv + t] = sqrtf(q0s);
  k0[(size_t)bh * Tv + t] = sqrtf(k0s);
}

// ---------------- causal hyperbolic flash attention ----------------
__global__ __launch_bounds__(256) void hyp_attn_kernel(
    const _Float16* __restrict__ Q, const _Float16* __restrict__ Kd,
    const _Float16* __restrict__ VT, const float* __restrict__ q0g,
    const float* __restrict__ k0g, const float* __restrict__ hyp,
    _Float16* __restrict__ out) {
  __shared__ _Float16 pshare[8][16][32];
  int bh = blockIdx.y;
  int h = bh & (Hv - 1), b = bh >> 4;
  int wave = threadIdx.x >> 5, lane = threadIdx.x & 31;
  int sg = lane >> 4, n = lane & 15;
  int qw = blockIdx.x * 128 + wave * 16;

  const _Float16* Qp = Q + ((size_t)bh * Tv + qw) * HDv;
  const _Float16* Kp = Kd + (size_t)bh * Tv * HDv;
  const _Float16* Vp = VT + (size_t)bh * HDv * Tv;
  const float* q0p = q0g + (size_t)bh * Tv + qw;
  const float* k0p = k0g + (size_t)bh * Tv;
  float kc = hyp[h];
  float sk = sqrtf(kc), rkc = 1.0f / kc;

  v16h aq0 = load_a_frag(Qp, HDv);        // d = 0..31
  v16h aq1 = load_a_frag(Qp + 32, HDv);   // d = 32..63
  float q0v[8], mrun[8], lrun[8];
#pragma unroll
  for (int j = 0; j < 8; ++j) {
    q0v[j] = q0p[j + 8 * sg];
    mrun[j] = -3.0e38f;
    lrun[j] = 0.f;
  }
  v8f oacc[4] = {};

  int sEnd = qw + 16;  // need keys < sEnd (causal)
  for (int sBase = 0; sBase < sEnd; sBase += 32) {
    v8f sacc[2];
#pragma unroll
    for (int c = 0; c < 2; ++c) {
      v16h bk0 = load_b_frag(Kp + (size_t)(sBase + c * 16) * HDv, HDv);
      v16h bk1 = load_b_frag(Kp + (size_t)(sBase + c * 16) * HDv + 32, HDv);
      v8f z = {};
      z = __builtin_amdgcn_wmma_f32_16x16x32_f16(false, aq0, false, bk0, (short)0, z, false, false);
      z = __builtin_amdgcn_wmma_f32_16x16x32_f16(false, aq1, false, bk1, (short)0, z, false, false);
      sacc[c] = z;
    }
    float p[2][8];
#pragma unroll
    for (int j = 0; j < 8; ++j) {
      int t = qw + j + 8 * sg;
      float mloc = -3.0e38f;
#pragma unroll
      for (int c = 0; c < 2; ++c) {
        int key = sBase + c * 16 + n;
        float l2 = sacc[c][j] - q0v[j] * k0p[key];
        float arg = fmaxf(-l2 * rkc, 1.0f + 1e-6f);
        float dis = sk * logf(arg + sqrtf(arg * arg - 1.0f));
        float sc = (key <= t) ? -dis : -3.0e38f;
        p[c][j] = sc;
        mloc = fmaxf(mloc, sc);
      }
#pragma unroll
      for (int o = 1; o < 16; o <<= 1) mloc = fmaxf(mloc, __shfl_xor(mloc, o, 32));
      float mnew = fmaxf(mrun[j], mloc);
      float alpha = expf(mrun[j] - mnew);
      float rs = 0.f;
#pragma unroll
      for (int c = 0; c < 2; ++c) {
        float pe = expf(p[c][j] - mnew);
        p[c][j] = pe;
        rs += pe;
      }
#pragma unroll
      for (int o = 1; o < 16; o <<= 1) rs += __shfl_xor(rs, o, 32);
      lrun[j] = lrun[j] * alpha + rs;
      mrun[j] = mnew;
#pragma unroll
      for (int dc = 0; dc < 4; ++dc) oacc[dc][j] *= alpha;
    }
    // stage P through per-wave LDS to re-lay as A-fragment (same-wave DS order)
#pragma unroll
    for (int c = 0; c < 2; ++c)
#pragma unroll
      for (int j = 0; j < 8; ++j)
        pshare[wave][j + 8 * sg][c * 16 + n] = (_Float16)p[c][j];
    v16h pf;
    {
      const _Float16* pr = &pshare[wave][n][8 * sg];
      v8h lo = *(const v8h*)pr;
      v8h hi = *(const v8h*)(pr + 16);
#pragma unroll
      for (int i = 0; i < 8; ++i) { pf[i] = lo[i]; pf[i + 8] = hi[i]; }
    }
#pragma unroll
    for (int dc = 0; dc < 4; ++dc) {
      v16h bv = load_b_frag(Vp + (size_t)(dc * 16) * Tv + sBase, Tv);
      oacc[dc] = __builtin_amdgcn_wmma_f32_16x16x32_f16(
          false, pf, false, bv, (short)0, oacc[dc], false, false);
    }
  }
#pragma unroll
  for (int dc = 0; dc < 4; ++dc)
#pragma unroll
    for (int j = 0; j < 8; ++j) {
      int t = qw + j + 8 * sg;
      float v = oacc[dc][j] / lrun[j];
      out[((size_t)(b * Tv + t)) * Cv + h * HDv + dc * 16 + n] = (_Float16)v;
    }
}

// ---------------- SiLU gating: g = u * silu(v) ----------------
__global__ __launch_bounds__(256) void silu_gate_kernel(
    const _Float16* __restrict__ uv, _Float16* __restrict__ g) {
  size_t i = (size_t)blockIdx.x * 256 + threadIdx.x;  // over NT*4C
  size_t m = i >> 12, nn = i & 4095;
  float u = (float)uv[m * (8 * Cv) + nn];
  float v = (float)uv[m * (8 * Cv) + 4 * Cv + nn];
  float s = v / (1.0f + expf(-v));
  g[i] = (_Float16)(u * s);
}

// ---------------- host orchestration ----------------
static inline size_t align256(size_t x) { return (x + 255) & ~(size_t)255; }

extern "C" void kernel_launch(void* const* d_in, const int* in_sizes, int n_in,
                              void* d_out, int out_size, void* d_ws, size_t ws_size,
                              hipStream_t stream) {
  (void)in_sizes; (void)n_in; (void)out_size; (void)ws_size;
  const float* x     = (const float*)d_in[0];
  const float* w_qkv = (const float*)d_in[1];
  const float* w_out = (const float*)d_in[2];
  const float* hyp   = (const float*)d_in[3];
  const float* w_uv  = (const float*)d_in[4];
  const float* b_uv  = (const float*)d_in[5];
  const float* w_mlp = (const float*)d_in[6];
  const float* b_mlp = (const float*)d_in[7];
  float* out = (float*)d_out;

  char* ws = (char*)d_ws;
  size_t off = 0;
  auto alloc = [&](size_t bytes) { char* p = ws + off; off = align256(off + bytes); return p; };

  _Float16* hA      = (_Float16*)alloc((size_t)NT * Cv * 2);
  _Float16* Wqkv16  = (_Float16*)alloc((size_t)3 * Cv * Cv * 2);
  _Float16* Wout16  = (_Float16*)alloc((size_t)Cv * Cv * 2);
  _Float16* Wuv16   = (_Float16*)alloc((size_t)8 * Cv * Cv * 2);
  _Float16* Wmlp16  = (_Float16*)alloc((size_t)Cv * 4 * Cv * 2);
  _Float16* qkv16   = (_Float16*)alloc((size_t)NT * 3 * Cv * 2);
  _Float16* Q16     = (_Float16*)alloc((size_t)Bv * Hv * Tv * HDv * 2);
  _Float16* K16     = (_Float16*)alloc((size_t)Bv * Hv * Tv * HDv * 2);
  _Float16* VT16    = (_Float16*)alloc((size_t)Bv * Hv * HDv * Tv * 2);
  float*    q0f     = (float*)alloc((size_t)Bv * Hv * Tv * 4);
  float*    k0f     = (float*)alloc((size_t)Bv * Hv * Tv * 4);
  _Float16* attn16  = (_Float16*)alloc((size_t)NT * Cv * 2);
  float*    x1      = (float*)alloc((size_t)NT * Cv * 4);
  _Float16* h2      = (_Float16*)alloc((size_t)NT * Cv * 2);
  _Float16* uv16    = (_Float16*)alloc((size_t)NT * 8 * Cv * 2);
  _Float16* g16     = (_Float16*)alloc((size_t)NT * 4 * Cv * 2);

  // 1) h = rmsnorm(x) in f16
  rmsnorm_f16_kernel<<<NT, 256, 0, stream>>>(x, hA);

  // 2) convert weights to f16
  auto conv = [&](const float* src, _Float16* dst, int n) {
    f32_to_f16_kernel<<<(n + 255) / 256, 256, 0, stream>>>(src, dst, n);
  };
  conv(w_qkv, Wqkv16, 3 * Cv * Cv);
  conv(w_out, Wout16, Cv * Cv);
  conv(w_uv,  Wuv16,  8 * Cv * Cv);
  conv(w_mlp, Wmlp16, Cv * 4 * Cv);

  // 3) qkv = h @ w_qkv^T   [NT x 3C]
  gemm_wmma_kernel<<<(NT / 128) * ((3 * Cv) / 128), 256, 0, stream>>>(
      hA, Wqkv16, NT, 3 * Cv, Cv, nullptr, nullptr, nullptr, qkv16);

  // 4) rotary + prefactors + V transpose
  rotary_pack_kernel<<<(NT * Hv) / 256, 256, 0, stream>>>(
      qkv16, hyp, Q16, K16, VT16, q0f, k0f);

  // 5) causal hyperbolic attention
  {
    dim3 grid(Tv / 128, Bv * Hv);
    hyp_attn_kernel<<<grid, 256, 0, stream>>>(Q16, K16, VT16, q0f, k0f, hyp, attn16);
  }

  // 6) x1 = x + attn @ w_out^T   (f32)
  gemm_wmma_kernel<<<(NT / 128) * (Cv / 128), 256, 0, stream>>>(
      attn16, Wout16, NT, Cv, Cv, nullptr, x, x1, nullptr);

  // 7) h2 = rmsnorm(x1)
  rmsnorm_f16_kernel<<<NT, 256, 0, stream>>>(x1, h2);

  // 8) uv = h2 @ w_uv^T + b_uv   [NT x 8C]
  gemm_wmma_kernel<<<(NT / 128) * ((8 * Cv) / 128), 256, 0, stream>>>(
      h2, Wuv16, NT, 8 * Cv, Cv, b_uv, nullptr, nullptr, uv16);

  // 9) g = u * silu(v)   [NT x 4C]
  silu_gate_kernel<<<((size_t)NT * 4 * Cv) / 256, 256, 0, stream>>>(uv16, g16);

  // 10) out = x1 + g @ w_mlp^T + b_mlp   (f32)
  gemm_wmma_kernel<<<(NT / 128) * (Cv / 128), 256, 0, stream>>>(
      g16, Wmlp16, NT, Cv, 4 * Cv, b_mlp, x1, out, nullptr);
}
